// GNNSparsifier_61100204753370
// MI455X (gfx1250) — matrix operand order repaired and checked by reading, not compiled, not run on hardware
//
#include <hip/hip_runtime.h>
#include <hip/hip_bf16.h>
#include <math.h>

typedef __attribute__((ext_vector_type(16))) _Float16 v16h;
typedef __attribute__((ext_vector_type(8)))  float    v8f;

// ---------------------------------------------------------------------------
// Intra-wave LDS producer->consumer fence (wave32 lanes are lock-step; we only
// need the DS counter drained; CDNA5 split wait counters).
// ---------------------------------------------------------------------------
__device__ __forceinline__ void lds_fence_wave() {
  __builtin_amdgcn_wave_barrier();
  asm volatile("s_wait_dscnt 0" ::: "memory");
  __builtin_amdgcn_wave_barrier();
}

// ---------------------------------------------------------------------------
// Utility kernels
// ---------------------------------------------------------------------------
__global__ void k_zero(float* __restrict__ p, int n) {
  int i = blockIdx.x * blockDim.x + threadIdx.x;
  if (i < n) p[i] = 0.0f;
}

__global__ void k_degree(const int* __restrict__ dst, float* __restrict__ deg, int E) {
  int e = blockIdx.x * blockDim.x + threadIdx.x;
  if (e < E) atomicAdd(&deg[dst[e]], 1.0f);
}

// One-off transpose+convert: Wt[n*K + k] = (f16) W[k*64 + n]   (W is [K,64])
__global__ void k_cvt_w(const float* __restrict__ W, _Float16* __restrict__ Wt, int K) {
  int i = blockIdx.x * blockDim.x + threadIdx.x;
  if (i < K * 64) {
    int n = i / K, k = i - n * K;
    Wt[i] = (_Float16)W[k * 64 + n];
  }
}

// Scatter h[src] into agg[dst] (segment_sum). One (edge, 4-dim chunk) per thread.
template <int D>
__global__ void k_scatter(const float* __restrict__ h, const int* __restrict__ src,
                          const int* __restrict__ dst, float* __restrict__ agg, int E) {
  const int CPE = D / 4;
  int t = blockIdx.x * blockDim.x + threadIdx.x;
  int e = t / CPE;
  int c = t - e * CPE;
  if (e >= E) return;
  int s = src[e], d = dst[e];
  const float4 v = *(const float4*)(h + (size_t)s * D + c * 4);
  float* out = agg + (size_t)d * D + c * 4;
  atomicAdd(out + 0, v.x);
  atomicAdd(out + 1, v.y);
  atomicAdd(out + 2, v.z);
  atomicAdd(out + 3, v.w);
}

// ---------------------------------------------------------------------------
// Node layer: hout = ReLU((hin + agg/deg) @ W + b), W pre-transposed f16 [64][K].
// Block = 128 threads = 4 waves; each wave computes 16x64 output tiles via
// v_wmma_f32_16x16x32_f16, grid-striding over tiles to amortize LDS staging.
// ---------------------------------------------------------------------------
template <int K>
__global__ __launch_bounds__(128) void k_node_layer(
    const float* __restrict__ hin, const float* __restrict__ agg,
    const float* __restrict__ deg, const _Float16* __restrict__ Wt,
    const float* __restrict__ bias, float* __restrict__ hout,
    _Float16* __restrict__ hout16, int nN) {
  __shared__ __align__(32) _Float16 Wlds[64 * K];      // Wlds[n][k]
  __shared__ float biasLds[64];
  __shared__ __align__(32) _Float16 Atile[4][16 * K];  // per-wave A tiles
  __shared__ float invLds[4][16];

  const int tid = threadIdx.x;
  const int lane = tid & 31;
  const int w = tid >> 5;

  for (int i = tid; i < (64 * K) / 8; i += 128)        // 128-bit copies
    ((uint4*)Wlds)[i] = ((const uint4*)Wt)[i];
  if (tid < 64) biasLds[tid] = bias[tid];
  __syncthreads();

  const int m = lane & 15;
  const int ncol = lane & 15;
  const int kadd = (lane < 16) ? 0 : 16;
  const int boff = (lane < 16) ? 0 : 16;
  const int mbase = (lane < 16) ? 0 : 8;
  const int ntiles = nN >> 4;  // N % 16 == 0

  for (int tile = blockIdx.x * 4 + w; tile < ntiles; tile += gridDim.x * 4) {
    const int base = tile * 16;

    if (lane < 16) invLds[w][lane] = 1.0f / fmaxf(deg[base + lane], 1.0f);
    lds_fence_wave();
    for (int i = lane; i < 16 * K; i += 32) {
      int r = i / K, k = i % K;
      size_t off = (size_t)(base + r) * K + k;
      Atile[w][i] = (_Float16)(hin[off] + agg[off] * invLds[w][r]);
    }
    lds_fence_wave();

    v8f acc[4];
#pragma unroll
    for (int nt = 0; nt < 4; ++nt) {
      float bv = biasLds[nt * 16 + ncol];
#pragma unroll
      for (int j = 0; j < 8; ++j) acc[nt][j] = bv;
    }

#pragma unroll
    for (int kb = 0; kb < K / 32; ++kb) {
      const int k0 = kb * 32;
      v16h a;
      const _Float16* ar = &Atile[w][m * K + k0 + kadd];
#pragma unroll
      for (int i = 0; i < 8; ++i) { a[i] = ar[i]; a[8 + i] = ar[8 + i]; }
#pragma unroll
      for (int nt = 0; nt < 4; ++nt) {
        const v16h b = *(const v16h*)&Wlds[(nt * 16 + ncol) * K + k0 + boff];
        acc[nt] = __builtin_amdgcn_wmma_f32_16x16x32_f16(
            false, a, false, b, (short)0, acc[nt], false, false);
      }
    }

#pragma unroll
    for (int nt = 0; nt < 4; ++nt) {
#pragma unroll
      for (int j = 0; j < 8; ++j) {
        float v = fmaxf(acc[nt][j], 0.0f);
        size_t o = (size_t)(base + mbase + j) * 64 + nt * 16 + ncol;
        hout[o] = v;
        if (hout16) hout16[o] = (_Float16)v;
      }
    }
  }
}

// ---------------------------------------------------------------------------
// Fused edge MLP: feat = [hs, hd, hs*hd, |hs-hd|] (256) built on the fly
// (never touches HBM); hidden = ReLU(feat @ We1 + be1); logit = hidden @ We2
// + be2; prob = sigmoid(logit). Grid-stride over 16-edge tiles; 32 WMMAs/tile.
// ---------------------------------------------------------------------------
__global__ __launch_bounds__(128) void k_edge_mlp(
    const _Float16* __restrict__ H16, const int* __restrict__ src,
    const int* __restrict__ dst, const _Float16* __restrict__ We1t,
    const float* __restrict__ be1, const float* __restrict__ We2,
    const float* __restrict__ be2, float* __restrict__ logits,
    float* __restrict__ probs, int E) {
  __shared__ __align__(32) _Float16 Wlds[64 * 256];  // Wlds[n][f]
  __shared__ float b1Lds[64], w2Lds[64];
  __shared__ __align__(16) _Float16 hsT[4][16 * 64];
  __shared__ __align__(16) _Float16 hdT[4][16 * 64];
  __shared__ float red[4][16 * 16];
  __shared__ int sIdx[4][16], dIdx[4][16];

  const int tid = threadIdx.x;
  const int lane = tid & 31;
  const int w = tid >> 5;

  for (int i = tid; i < (64 * 256) / 8; i += 128)    // 32 KB in 128-bit copies
    ((uint4*)Wlds)[i] = ((const uint4*)We1t)[i];
  if (tid < 64) { b1Lds[tid] = be1[tid]; w2Lds[tid] = We2[tid]; }
  __syncthreads();

  const int m = lane & 15;
  const int ncol = lane & 15;
  const int kadd = (lane < 16) ? 0 : 16;
  const int boff = (lane < 16) ? 0 : 16;
  const int mbase = (lane < 16) ? 0 : 8;
  const float bias2 = be2[0];
  const int ntiles = E >> 4;  // E % 16 == 0

  for (int tile = blockIdx.x * 4 + w; tile < ntiles; tile += gridDim.x * 4) {
    const int base = tile * 16;

    if (lane < 16) {
      sIdx[w][lane] = src[base + lane];
      dIdx[w][lane] = dst[base + lane];
    }
    lds_fence_wave();

    // Gather 16 h_src / h_dst rows (64 f16 = 128 B = 8 uint4 each) into LDS.
    for (int i = lane; i < 16 * 8; i += 32) {
      int r = i >> 3, c = i & 7;
      ((uint4*)&hsT[w][r * 64])[c] =
          ((const uint4*)(H16 + (size_t)sIdx[w][r] * 64))[c];
      ((uint4*)&hdT[w][r * 64])[c] =
          ((const uint4*)(H16 + (size_t)dIdx[w][r] * 64))[c];
    }
    lds_fence_wave();

    const _Float16* hs = &hsT[w][m * 64];
    const _Float16* hd = &hdT[w][m * 64];

    v8f acc[4];
#pragma unroll
    for (int nt = 0; nt < 4; ++nt) {
      float bv = b1Lds[nt * 16 + ncol];
#pragma unroll
      for (int j = 0; j < 8; ++j) acc[nt][j] = bv;
    }

#pragma unroll
    for (int kb = 0; kb < 8; ++kb) {
      const int k0 = kb * 32;
      v16h a;
#pragma unroll
      for (int half = 0; half < 2; ++half) {
        const int fb = k0 + kadd + half * 8;  // 8-run never crosses a segment
        const int seg = fb >> 6;
        const int db = fb & 63;
#pragma unroll
        for (int i = 0; i < 8; ++i) {
          const int d = db + i;
          _Float16 v;
          if (seg == 0)      v = hs[d];
          else if (seg == 1) v = hd[d];
          else if (seg == 2) v = (_Float16)((float)hs[d] * (float)hd[d]);
          else               v = (_Float16)fabsf((float)hs[d] - (float)hd[d]);
          a[half * 8 + i] = v;
        }
      }
#pragma unroll
      for (int nt = 0; nt < 4; ++nt) {
        const v16h b = *(const v16h*)&Wlds[(nt * 16 + ncol) * 256 + k0 + boff];
        acc[nt] = __builtin_amdgcn_wmma_f32_16x16x32_f16(
            false, a, false, b, (short)0, acc[nt], false, false);
      }
    }

    // Second layer: ReLU then dot with We2; per-lane partials, reduce via LDS.
    float s[8];
#pragma unroll
    for (int j = 0; j < 8; ++j) s[j] = 0.0f;
#pragma unroll
    for (int nt = 0; nt < 4; ++nt) {
      const float w2 = w2Lds[nt * 16 + ncol];
#pragma unroll
      for (int j = 0; j < 8; ++j) s[j] += fmaxf(acc[nt][j], 0.0f) * w2;
    }
#pragma unroll
    for (int j = 0; j < 8; ++j) red[w][(mbase + j) * 16 + ncol] = s[j];
    lds_fence_wave();

    if (lane < 16) {
      float t = bias2;
#pragma unroll
      for (int c = 0; c < 16; ++c) t += red[w][lane * 16 + c];
      const int e = base + lane;
      logits[e] = t;
      probs[e] = 1.0f / (1.0f + expf(-t));
    }
    lds_fence_wave();  // red[] reused next iteration
  }
}

// ---------------------------------------------------------------------------
// Host-side launch
// ---------------------------------------------------------------------------
static inline size_t align256(size_t x) { return (x + 255) & ~(size_t)255; }

extern "C" void kernel_launch(void* const* d_in, const int* in_sizes, int n_in,
                              void* d_out, int out_size, void* d_ws, size_t ws_size,
                              hipStream_t stream) {
  const float* x   = (const float*)d_in[0];
  const float* W1  = (const float*)d_in[1];
  const float* b1  = (const float*)d_in[2];
  const float* W2  = (const float*)d_in[3];
  const float* b2  = (const float*)d_in[4];
  const float* We1 = (const float*)d_in[5];
  const float* be1 = (const float*)d_in[6];
  const float* We2 = (const float*)d_in[7];
  const float* be2 = (const float*)d_in[8];
  const int*   ei  = (const int*)d_in[9];

  const int N = in_sizes[0] / 128;  // 50000
  const int E = in_sizes[9] / 2;    // 800000
  const int* src = ei;
  const int* dst = ei + E;

  // Workspace layout
  char* ws = (char*)d_ws;
  size_t off = 0;
  float* deg = (float*)(ws + off); off += align256((size_t)N * 4);
  float* agg = (float*)(ws + off); off += align256((size_t)N * 128 * 4);
  float* h1  = (float*)(ws + off); off += align256((size_t)N * 64 * 4);
  _Float16* H16  = (_Float16*)(ws + off); off += align256((size_t)N * 64 * 2);
  _Float16* W1t  = (_Float16*)(ws + off); off += align256((size_t)64 * 128 * 2);
  _Float16* W2t  = (_Float16*)(ws + off); off += align256((size_t)64 * 64 * 2);
  _Float16* We1t = (_Float16*)(ws + off); off += align256((size_t)64 * 256 * 2);
  (void)ws_size; (void)n_in; (void)out_size;

  float* Hout   = (float*)d_out;           // [N,64]
  float* logits = Hout + (size_t)N * 64;   // [E]
  float* probs  = logits + E;              // [E]

  const int TB = 256;
  auto blks = [](long long n, int tb) { return (int)((n + tb - 1) / tb); };

  // One-off weight transpose + f16 conversion
  k_cvt_w<<<blks(64 * 128, TB), TB, 0, stream>>>(W1, W1t, 128);
  k_cvt_w<<<blks(64 * 64, TB), TB, 0, stream>>>(W2, W2t, 64);
  k_cvt_w<<<blks(64 * 256, TB), TB, 0, stream>>>(We1, We1t, 256);

  // ---- Layer 1 ----
  k_zero<<<blks(N, TB), TB, 0, stream>>>(deg, N);
  k_zero<<<blks((long long)N * 128, TB), TB, 0, stream>>>(agg, N * 128);
  k_degree<<<blks(E, TB), TB, 0, stream>>>(dst, deg, E);
  k_scatter<128><<<blks((long long)E * 32, TB), TB, 0, stream>>>(x, src, dst, agg, E);
  {
    int tiles = N / 16;
    int grid = (tiles + 3) / 4; if (grid > 512) grid = 512;
    k_node_layer<128><<<grid, 128, 0, stream>>>(x, agg, deg, W1t, b1, h1, nullptr, N);
  }

  // ---- Layer 2 ----
  k_zero<<<blks((long long)N * 64, TB), TB, 0, stream>>>(agg, N * 64);
  k_scatter<64><<<blks((long long)E * 16, TB), TB, 0, stream>>>(h1, src, dst, agg, E);
  {
    int tiles = N / 16;
    int grid = (tiles + 3) / 4; if (grid > 512) grid = 512;
    k_node_layer<64><<<grid, 128, 0, stream>>>(h1, agg, deg, W2t, b2, Hout, H16, N);
  }

  // ---- Fused edge MLP ----
  {
    int tiles = E / 16;                     // 50000 tiles
    int grid = (tiles + 3) / 4; if (grid > 1024) grid = 1024;
    k_edge_mlp<<<grid, 128, 0, stream>>>(H16, src, dst, We1t, be1, We2, be2,
                                         logits, probs, E);
  }
}
